// EMCAD_35845797052919
// MI455X (gfx1250) — matrix operand order repaired and checked by reading, not compiled, not run on hardware
//
#include <hip/hip_runtime.h>

// ---------------------------------------------------------------------------
// EMCAD block (DySample + EUCB2) for MI455X / gfx1250, wave32.
//  k_off1 : 1x1 conv 128->32 as fp32 WMMA GEMM (v_wmma_f32_16x16x4_f32)
//  k_off2 : 3x3 dilated(2) conv 32->32 + /G + init_pos  (VALU, L2-resident)
//  k_fused: grid-sample -> LDS(f16) -> depthwise+BN+ReLU -> LDS ->
//           1x1 conv 128->64 as f16 WMMA GEMM (v_wmma_f32_16x16x32_f16),
//           with async global->LDS DMA staging of the depthwise weights.
// ---------------------------------------------------------------------------

typedef float    v2f  __attribute__((ext_vector_type(2)));
typedef float    v8f  __attribute__((ext_vector_type(8)));
typedef _Float16 v16h __attribute__((ext_vector_type(16)));
typedef _Float16 v8h  __attribute__((ext_vector_type(8)));
typedef int      v4i  __attribute__((vector_size(16)));   // b128 packet

#define BB    16
#define CC    128
#define HH    96
#define WW    96
#define GG    4
#define CO    32
#define COUT  64
#define H2    192
#define W2    192
#define HWIN  (HH * WW)
#define PI_F  3.14159265358979f

#if __has_builtin(__builtin_amdgcn_global_load_async_to_lds_b128) && \
    __has_builtin(__builtin_amdgcn_s_wait_asynccnt)
#define USE_ASYNC_LDS 1
#endif

// ========================= kernel 1: 1x1 conv (WMMA f32) ====================
// GEMM: D[32 co x 16 px] per wave, K=128 in steps of 4 (V_WMMA_F32_16X16X4_F32)
// A layout (32-bit A 16x4, ISA 7.12.2): lanes0-15 VGPR0=K0,VGPR1=K1;
//                                       lanes16-31 VGPR0=K2,VGPR1=K3.
// B layout (4x16): VGPR0 lanes0-15 = row K0, lanes16-31 = row K1; VGPR1 = K2,K3.
// C/D f32 16x16: lane L, VGPR r -> M = r + 8*(L>>4), N = L&15.
__global__ __launch_bounds__(256) void k_off1(const float* __restrict__ x,
                                              const float* __restrict__ Wo1,
                                              const float* __restrict__ bo1,
                                              float* __restrict__ off1)
{
  const int lane = threadIdx.x & 31;
  const int wave = threadIdx.x >> 5;
  const int m    = lane & 15;
  const int hi   = lane >> 4;
  const int tile = blockIdx.x * 8 + wave;      // one 16-pixel tile per wave
  const int pxB  = tile * 16;
  const int b    = pxB / HWIN;                 // HWIN % 16 == 0 -> uniform b
  const int hw   = (pxB % HWIN) + m;           // this lane's pixel (N = m)
  const float* xb = x + (size_t)b * CC * HWIN;

  v8f acc0 = {}; v8f acc1 = {};
  for (int k = 0; k < CC / 4; ++k) {
    const int kc = k * 4;
    v2f a0, a1, bv;
    a0.x = Wo1[m * CC + kc + 2 * hi];          // co tile 0 (co 0..15)
    a0.y = Wo1[m * CC + kc + 2 * hi + 1];
    a1.x = Wo1[(m + 16) * CC + kc + 2 * hi];   // co tile 1 (co 16..31)
    a1.y = Wo1[(m + 16) * CC + kc + 2 * hi + 1];
    bv.x = xb[(kc + hi) * HWIN + hw];          // activations, NCHW strided
    bv.y = xb[(kc + 2 + hi) * HWIN + hw];
#if __has_builtin(__builtin_amdgcn_wmma_f32_16x16x4_f32)
    acc0 = __builtin_amdgcn_wmma_f32_16x16x4_f32(false, a0, false, bv,
                                                 (short)0, acc0, false, false);
    acc1 = __builtin_amdgcn_wmma_f32_16x16x4_f32(false, a1, false, bv,
                                                 (short)0, acc1, false, false);
#else
    // placeholder so the TU still compiles if the f32 WMMA builtin is absent
    acc0[0] += a0.x * bv.x;  acc1[0] += a1.x * bv.x;
#endif
  }
  float* ob = off1 + (size_t)b * CO * HWIN;
  #pragma unroll
  for (int r = 0; r < 8; ++r) {
    const int co = r + 8 * hi;
    ob[co * HWIN + hw]        = acc0[r] + bo1[co];
    ob[(co + 16) * HWIN + hw] = acc1[r] + bo1[co + 16];
  }
}

// ================ kernel 2: 3x3 dilated conv + /G + init_pos ================
// 38 MB working set -> fully L2-resident on a 192 MB L2; VALU is fine.
__global__ __launch_bounds__(256) void k_off2(const float* __restrict__ off1,
                                              const float* __restrict__ Wo2,
                                              float* __restrict__ offs)
{
  const int idx = blockIdx.x * 256 + threadIdx.x;
  const int w  = idx % WW;
  const int h  = (idx / WW) % HH;
  const int co = (idx / HWIN) % CO;
  const int b  = idx / (HWIN * CO);

  const float* ib = off1 + (size_t)b * CO * HWIN;
  const float* wk = Wo2 + co * CO * 9;
  float acc = 0.f;
  for (int ci = 0; ci < CO; ++ci) {
    const float* ic = ib + ci * HWIN;
    const float* wc = wk + ci * 9;
    #pragma unroll
    for (int ky = 0; ky < 3; ++ky) {
      const int hh2 = h + 2 * ky - 2;                // dilation 2, pad 2
      if ((unsigned)hh2 >= HH) continue;
      #pragma unroll
      for (int kx = 0; kx < 3; ++kx) {
        const int ww2 = w + 2 * kx - 2;
        if ((unsigned)ww2 >= WW) continue;
        acc += ic[hh2 * WW + ww2] * wc[ky * 3 + kx];
      }
    }
  }
  // init_pos(co): d = co>>4 (x/y), value depends on (row%S, col) of pixel
  // shuffle cell; reduces to +/-0.25 per the reference's meshgrid/transpose.
  const int d = co >> 4, rbit = (co >> 1) & 1, cbit = co & 1;
  const float ip = (d == 0) ? (cbit ? 0.25f : -0.25f)
                            : (rbit ? 0.25f : -0.25f);
  offs[idx] = acc * 0.25f + ip;                      // off/G + init_pos
}

// ===== kernel 3: fused DySample + depthwise3x3 + BN + ReLU + 1x1 WMMA =======
#define TPX  32            // output pixels per tile (along W2)
#define TCOL (TPX + 2)     // +2 halo cols for the depthwise 3x3

__global__ __launch_bounds__(256) void k_fused(
    const float* __restrict__ x,    const float* __restrict__ offs,
    const float* __restrict__ Wdw,  const float* __restrict__ gamma,
    const float* __restrict__ beta, const float* __restrict__ mean,
    const float* __restrict__ var,  const float* __restrict__ Wpw,
    const float* __restrict__ bpw,  float* __restrict__ out)
{
  __shared__ float                  s_ix[3][GG][TCOL];   // sample coords
  __shared__ float                  s_iy[3][GG][TCOL];
  __shared__ _Float16               s_up[CC][3][TCOL];   // sampled halo tile
  __shared__ __align__(32) _Float16 s_ys[TPX][CC];       // relu(bn(dw)) f16
  __shared__ __align__(32) _Float16 s_wpw[COUT * CC];    // 1x1 weights f16
  __shared__ float                  s_scale[CC], s_shift[CC];
  __shared__ __align__(16) float    s_wdw[CC * 9];       // depthwise weights

  const int tid = threadIdx.x;
  const int wt  = blockIdx.x % (W2 / TPX);
  const int ho  = (blockIdx.x / (W2 / TPX)) % H2;
  const int b   = blockIdx.x / ((W2 / TPX) * H2);
  const int wo0 = wt * TPX;

  // ---- phase 0: stage weights / BN params into LDS ----
#if USE_ASYNC_LDS
  // gfx1250 async DMA: global -> LDS without a VGPR round-trip, overlapped
  // with the coordinate math below; settled by s_wait_asynccnt + barrier.
  for (int i = tid * 4; i < CC * 9; i += 256 * 4) {      // 288 b128 packets
    __builtin_amdgcn_global_load_async_to_lds_b128(
        (v4i*)(Wdw + i), (v4i*)&s_wdw[i], 0, 0);
  }
#else
  for (int i = tid; i < CC * 9; i += 256) s_wdw[i] = Wdw[i];
#endif
  for (int i = tid; i < COUT * CC; i += 256) s_wpw[i] = (_Float16)Wpw[i];
  if (tid < CC) {
    const float sc = gamma[tid] * rsqrtf(var[tid] + 1e-5f);
    s_scale[tid] = sc;
    s_shift[tid] = beta[tid] - mean[tid] * sc;
  }

  // ---- phase 1: sampling coordinates for the 3 x TCOL halo, per group ----
  // grid_sample(border, align_corners=False) collapses to
  //   ix = clamp(cw[w] + off_x - 0.5, 0, W-1)  (same for y)
  const float* offb = offs + (size_t)b * CO * HWIN;
  for (int i = tid; i < 3 * GG * TCOL; i += 256) {
    const int cc = i % TCOL;
    const int g  = (i / TCOL) % GG;
    const int rr = i / (TCOL * GG);
    const int hr = ho + rr - 1;
    const int wg = wo0 - 1 + cc;
    float ix = 0.f, iy = 0.f;
    if ((unsigned)hr < H2 && (unsigned)wg < W2) {
      const int h = hr >> 1, si = hr & 1;
      const int w = wg >> 1, sj = wg & 1;
      const int j = g * 4 + si * 2 + sj;        // pixel-shuffle channel
      const float cx = offb[(j        * HH + h) * WW + w];
      const float cy = offb[((16 + j) * HH + h) * WW + w];
      const float cw = (float)w + __sinf(PI_F * (w + 1) / WW);
      const float ch = (float)h + __sinf(PI_F * (h + 1) / HH);
      ix = fminf(fmaxf(cw + cx - 0.5f, 0.f), (float)(WW - 1));
      iy = fminf(fmaxf(ch + cy - 0.5f, 0.f), (float)(HH - 1));
    }
    s_ix[rr][g][cc] = ix;
    s_iy[rr][g][cc] = iy;
  }
#if USE_ASYNC_LDS
  __builtin_amdgcn_s_wait_asynccnt(0);          // DMA landed in s_wdw
#endif
  __syncthreads();

  // ---- phase 2: bilinear gather from x into s_up (f16) ----
  const float* xb = x + (size_t)b * CC * HWIN;
  for (int i = tid; i < CC * 3 * TCOL; i += 256) {
    const int cc = i % TCOL;
    const int rr = (i / TCOL) % 3;
    const int c  = i / (TCOL * 3);
    const int hr = ho + rr - 1;
    const int wg = wo0 - 1 + cc;
    float v = 0.f;                               // zero-pad for depthwise
    if ((unsigned)hr < H2 && (unsigned)wg < W2) {
      const int g = c >> 5;                      // channel group (C/G = 32)
      const float ix = s_ix[rr][g][cc];
      const float iy = s_iy[rr][g][cc];
      const int  x0 = (int)ix,  y0 = (int)iy;
      const float fx = ix - x0, fy = iy - y0;
      const int  x1 = min(x0 + 1, WW - 1), y1 = min(y0 + 1, HH - 1);
      const float* xc = xb + (size_t)c * HWIN;
      const float v00 = xc[y0 * WW + x0], v01 = xc[y0 * WW + x1];
      const float v10 = xc[y1 * WW + x0], v11 = xc[y1 * WW + x1];
      v = (v00 * (1.f - fx) + v01 * fx) * (1.f - fy)
        + (v10 * (1.f - fx) + v11 * fx) * fy;
    }
    s_up[c][rr][cc] = (_Float16)v;
  }
  __syncthreads();

  // ---- phase 3: depthwise 3x3 + BN + ReLU -> s_ys[px][c] (f16, B-operand) --
  for (int i = tid; i < CC * TPX; i += 256) {
    const int c = i % CC;                        // contiguous c per wave
    const int j = i / CC;                        // pixel within tile
    float acc = 0.f;
    #pragma unroll
    for (int rr = 0; rr < 3; ++rr)
      #pragma unroll
      for (int dd = 0; dd < 3; ++dd)
        acc += (float)s_up[c][rr][j + dd] * s_wdw[c * 9 + rr * 3 + dd];
    float yv = acc * s_scale[c] + s_shift[c];
    yv = fmaxf(yv, 0.f);                         // channel_shuffle == identity
    s_ys[j][c] = (_Float16)yv;
  }
  __syncthreads();

  // ---- phase 4: 1x1 conv 128->64 as f16 WMMA GEMM ----
  // 8 subtiles (4 co-tiles x 2 px-tiles), one per wave; K = 128 in 4 steps.
  // A f16 16x32, k order used here (same as B): lane m=L&15, K(e) = (L>>4)*16 + e
  //   -> two contiguous 8-half runs per lane from s_wpw.
  // B f16 32x16, same k order as A: lane L -> N = L&15, K = (L>>4)*16 + e
  //   -> 16 contiguous halves per lane from s_ys.
  const int lane = tid & 31;
  const int wave = tid >> 5;
  const int m    = lane & 15;
  const int hi   = lane >> 4;
  const int coT  = wave >> 1;                    // 0..3
  const int pxT  = wave & 1;                     // 0..1
  v8f acc = {};
  #pragma unroll
  for (int kb = 0; kb < CC / 32; ++kb) {
    const _Float16* ap = &s_wpw[(coT * 16 + m) * CC + kb * 32];
    const v8h lo = *(const v8h*)(ap + hi * 16);
    const v8h hh = *(const v8h*)(ap + hi * 16 + 8);
    v16h a;
    #pragma unroll
    for (int e = 0; e < 8; ++e) { a[e] = lo[e]; a[e + 8] = hh[e]; }
    const v16h bv = *(const v16h*)(&s_ys[pxT * 16 + m][kb * 32 + hi * 16]);
    acc = __builtin_amdgcn_wmma_f32_16x16x32_f16(false, a, false, bv,
                                                 (short)0, acc, false, false);
  }
  #pragma unroll
  for (int r = 0; r < 8; ++r) {
    const int co = coT * 16 + r + 8 * hi;
    const int wo = wo0 + pxT * 16 + m;
    out[(((size_t)b * COUT + co) * H2 + ho) * W2 + wo] = acc[r] + bpw[co];
  }
}

// ============================== launcher ====================================
extern "C" void kernel_launch(void* const* d_in, const int* in_sizes, int n_in,
                              void* d_out, int out_size, void* d_ws,
                              size_t ws_size, hipStream_t stream)
{
  const float* x    = (const float*)d_in[0];
  const float* Wo1  = (const float*)d_in[1];
  const float* bo1  = (const float*)d_in[2];
  const float* Wo2  = (const float*)d_in[3];
  const float* Wdw  = (const float*)d_in[4];
  const float* bn_g = (const float*)d_in[5];
  const float* bn_b = (const float*)d_in[6];
  const float* bn_m = (const float*)d_in[7];
  const float* bn_v = (const float*)d_in[8];
  const float* Wpw  = (const float*)d_in[9];
  const float* bpw  = (const float*)d_in[10];
  float* out  = (float*)d_out;

  float* off1 = (float*)d_ws;                          // [B,32,96,96] 18.9 MB
  float* offs = off1 + (size_t)BB * CO * HWIN;         // [B,32,96,96] 18.9 MB

  // 9216 pixel-tiles of 16, 8 waves/block
  k_off1<<<(BB * HWIN / 16) / 8, 256, 0, stream>>>(x, Wo1, bo1, off1);
  // one thread per offset element (exact multiple of 256)
  k_off2<<<(BB * CO * HWIN) / 256, 256, 0, stream>>>(off1, Wo2, offs);
  // one block per (b, ho, 32-px strip)
  k_fused<<<BB * H2 * (W2 / TPX), 256, 0, stream>>>(
      x, offs, Wdw, bn_g, bn_b, bn_m, bn_v, Wpw, bpw, out);
}
